// HHGNN_Poincare_Adaptive_17927193494053
// MI455X (gfx1250) — matrix operand order repaired and checked by reading, not compiled, not run on hardware
//
#include <hip/hip_runtime.h>
#include <math.h>

typedef _Float16 h8   __attribute__((ext_vector_type(8)));
typedef _Float16 v16h __attribute__((ext_vector_type(16)));
typedef float    v8f  __attribute__((ext_vector_type(8)));

#define NEG_SLOPE 0.2f

__device__ __forceinline__ unsigned fflip(float f) {
  unsigned u = __float_as_uint(f);
  unsigned mask = (unsigned)(-(int)(u >> 31)) | 0x80000000u;
  return u ^ mask;  // monotonic float -> uint mapping
}
__device__ __forceinline__ float funflip(unsigned u) {
  unsigned mask = (u >> 31) ? 0x80000000u : 0xFFFFFFFFu;
  return __uint_as_float(u ^ mask);
}

// ---------------- kernel: zero fill (graph-capture-safe init) ----------------
__global__ void k_fill_zero(unsigned* __restrict__ p, long long n) {
  long long i = (long long)blockIdx.x * blockDim.x + threadIdx.x;
  long long stride = (long long)gridDim.x * blockDim.x;
  for (; i < n; i += stride) p[i] = 0u;
}

// ---------------- kernel 1: TypedLinear via WMMA ----------------
// X0[n,:] = X[n,:] @ W[node_type[n]]   (grid.y = type t; rows with other types skipped)
__global__ void __launch_bounds__(256)
k_typed_linear(const float* __restrict__ X, const float* __restrict__ W,
               const int* __restrict__ ntype, float* __restrict__ X0, int N) {
  __shared__ __align__(16) _Float16 sA[16][136];    // X tile, f16, padded rows
  __shared__ __align__(16) _Float16 sWT[128][136];  // W_t transposed: [n][k]
  __shared__ int sTy[16];

  const int tid = threadIdx.x;
  const int r0  = blockIdx.x * 16;
  const int t   = blockIdx.y;
  const float* Wt = W + (size_t)t * 128 * 128;

  for (int j = tid; j < 16 * 128; j += 256) {       // X tile -> f16
    int row = j >> 7, col = j & 127;
    int g = r0 + row;
    float v = (g < N) ? X[(size_t)g * 128 + col] : 0.0f;
    sA[row][col] = (_Float16)v;
  }
  for (int j = tid; j < 128 * 128; j += 256) {      // W_t -> f16 transposed
    int k = j >> 7, n = j & 127;
    sWT[n][k] = (_Float16)Wt[j];
  }
  if (tid < 16) {
    int g = r0 + tid;
    sTy[tid] = (g < N) ? ntype[g] : -1;
  }
  __syncthreads();

  const int lane = tid & 31;
  const int n0   = (tid >> 5) * 16;  // 8 waves cover 128 output cols
  const int lo   = lane & 15;
  const int hi   = lane >> 4;

  v8f acc = {};
#pragma unroll
  for (int kb = 0; kb < 4; ++kb) {
    // A fragment (16x32 f16): lanes 0-15 hold K{0-7,16-23}, lanes 16-31 K{8-15,24-31}
    const h8* pa = (const h8*)&sA[lo][kb * 32 + hi * 8];
    h8 a0 = pa[0];
    h8 a1 = pa[2];  // +16 halves
    v16h a = __builtin_shufflevector(a0, a1, 0,1,2,3,4,5,6,7,8,9,10,11,12,13,14,15);
    // B fragment (32x16 f16): lanes 0-15 hold K{0-7,16-23}, lanes 16-31 K{8-15,24-31}, col = lane&15
    const h8* pb = (const h8*)&sWT[n0 + lo][kb * 32 + hi * 8];
    h8 b0 = pb[0];
    h8 b1 = pb[2];
    v16h b = __builtin_shufflevector(b0, b1, 0,1,2,3,4,5,6,7,8,9,10,11,12,13,14,15);
    acc = __builtin_amdgcn_wmma_f32_16x16x32_f16(false, a, false, b,
                                                 (short)0, acc, false, false);
  }
#pragma unroll
  for (int r = 0; r < 8; ++r) {       // D: VGPR r -> row r + 8*hi, col = lane&15
    int row = r + hi * 8;
    int g = r0 + row;
    if (g < N && sTy[row] == t)
      X0[(size_t)g * 128 + n0 + lo] = acc[r];
  }
}

// ---------------- kernel 2/6: attention logits + segment max ----------------
// One wave per incidence; lane loads float4 (4 channels); 4 lanes = 1 head.
__global__ void __launch_bounds__(256)
k_logits_max(const float* __restrict__ feat,     // rows gathered via gidx
             const int* __restrict__ gidx,       // row index per incidence
             const int* __restrict__ sidx,       // segment index per incidence
             const int* __restrict__ cls_of_seg, // class lookup array
             const int* __restrict__ cls_key,    // index into cls_of_seg (gidx or sidx)
             const float* __restrict__ att,      // [4,8,16]
             float* __restrict__ logit,          // [nnz,8]
             unsigned* __restrict__ mx,          // [nseg,8] flipped keys
             int nnz, int apply_relu_on_feat) {
  int gw = (int)(((long long)blockIdx.x * blockDim.x + threadIdx.x) >> 5);
  if (gw >= nnz) return;
  int lane = threadIdx.x & 31;
  int g = gidx[gw];
  int s = sidx[gw];
  int c = cls_of_seg[cls_key[gw]];
  float4 x4 = ((const float4*)(feat + (size_t)g * 128))[lane];
  if (apply_relu_on_feat) {
    x4.x = fmaxf(x4.x, 0.0f); x4.y = fmaxf(x4.y, 0.0f);
    x4.z = fmaxf(x4.z, 0.0f); x4.w = fmaxf(x4.w, 0.0f);
  }
  float4 a4 = ((const float4*)(att + (size_t)c * 128))[lane];
  float p = x4.x * a4.x + x4.y * a4.y + x4.z * a4.z + x4.w * a4.w;
  p += __shfl_xor(p, 1, 32);
  p += __shfl_xor(p, 2, 32);                 // 4-lane (one head) reduction
  float b = (p > 0.0f) ? p : NEG_SLOPE * p;  // leaky relu
  if ((lane & 3) == 0) {
    int h = lane >> 2;
    logit[(size_t)gw * 8 + h] = b;
    atomicMax(&mx[(size_t)s * 8 + h], fflip(b));
  }
}

// ---------------- kernel 3/7: exp + segment sum ----------------
__global__ void __launch_bounds__(256)
k_expsum(const int* __restrict__ sidx, float* __restrict__ logit,
         const unsigned* __restrict__ mx, float* __restrict__ sm, int nnz) {
  long long idx = (long long)blockIdx.x * blockDim.x + threadIdx.x;
  if (idx >= (long long)nnz * 8) return;
  int i = (int)(idx >> 3);
  int h = (int)(idx & 7);
  int s = sidx[i];
  float m = funflip(mx[(size_t)s * 8 + h]);
  if (!(m > -1e38f && m < 1e38f)) m = 0.0f;  // empty segment / NaN guard
  float ex = __expf(logit[idx] - m);
  logit[idx] = ex;
  atomicAdd(&sm[(size_t)s * 8 + h], ex);
}

// ---------------- kernel 4/8: weighted segment scatter-add ----------------
__global__ void __launch_bounds__(256)
k_weighted_accum(const float* __restrict__ feat, const int* __restrict__ gidx,
                 const int* __restrict__ sidx, const float* __restrict__ logit,
                 const float* __restrict__ sm, float* __restrict__ dst,
                 int nnz, int apply_relu_on_feat) {
  int gw = (int)(((long long)blockIdx.x * blockDim.x + threadIdx.x) >> 5);
  if (gw >= nnz) return;
  int lane = threadIdx.x & 31;
  int g = gidx[gw];
  int s = sidx[gw];
  int h = lane >> 2;
  float w = logit[(size_t)gw * 8 + h] / (sm[(size_t)s * 8 + h] + 1e-16f);
  float4 x4 = ((const float4*)(feat + (size_t)g * 128))[lane];
  if (apply_relu_on_feat) {
    x4.x = fmaxf(x4.x, 0.0f); x4.y = fmaxf(x4.y, 0.0f);
    x4.z = fmaxf(x4.z, 0.0f); x4.w = fmaxf(x4.w, 0.0f);
  }
  float* d = dst + (size_t)s * 128 + lane * 4;
  atomicAdd(d + 0, x4.x * w);
  atomicAdd(d + 1, x4.y * w);
  atomicAdd(d + 2, x4.z * w);
  atomicAdd(d + 3, x4.w * w);
}

extern "C" void kernel_launch(void* const* d_in, const int* in_sizes, int n_in,
                              void* d_out, int out_size, void* d_ws, size_t ws_size,
                              hipStream_t stream) {
  const float* X      = (const float*)d_in[0];
  const float* W      = (const float*)d_in[1];
  const float* att_e  = (const float*)d_in[2];
  const float* att_v  = (const float*)d_in[3];
  const int*   ntype  = (const int*)d_in[4];
  const int*   etype  = (const int*)d_in[5];
  const int*   vertex = (const int*)d_in[6];
  const int*   edges  = (const int*)d_in[7];
  float* out = (float*)d_out;

  const int N   = in_sizes[0] / 128;
  const int E   = in_sizes[5];
  const int nnz = in_sizes[6];

  // workspace layout
  float*    X0    = (float*)d_ws;                         // N*128
  float*    logit = X0 + (size_t)N * 128;                 // nnz*8 (beta, reused as alpha)
  unsigned* maxE  = (unsigned*)(logit + (size_t)nnz * 8); // E*8   ---- zeroed region start
  float*    sumE  = (float*)(maxE + (size_t)E * 8);       // E*8
  float*    Xe    = sumE + (size_t)E * 8;                 // E*128
  unsigned* maxV  = (unsigned*)(Xe + (size_t)E * 128);    // N*8
  float*    sumV  = (float*)(maxV + (size_t)N * 8);       // N*8   ---- zeroed region end
  long long zeroN = (long long)E * 16 + (long long)E * 128 + (long long)N * 16;

  // init accumulators (flipped-key 0 == "below every real float")
  k_fill_zero<<<(int)((zeroN + 255) / 256), 256, 0, stream>>>(maxE, zeroN);
  k_fill_zero<<<(out_size + 255) / 256, 256, 0, stream>>>((unsigned*)out, (long long)out_size);

  // 1) TypedLinear (WMMA)
  dim3 g1((N + 15) / 16, 4);
  k_typed_linear<<<g1, 256, 0, stream>>>(X, W, ntype, X0, N);

  const int waveBlocks = (nnz + 7) / 8;          // 8 waves / 256-thread block
  const int nhBlocks   = (int)(((long long)nnz * 8 + 255) / 256);

  // 2-4) edge-side attention: softmax over hyperedge segments, Xe = segsum
  k_logits_max<<<waveBlocks, 256, 0, stream>>>(X0, vertex, edges, etype, edges,
                                               att_e, logit, maxE, nnz, 0);
  k_expsum<<<nhBlocks, 256, 0, stream>>>(edges, logit, maxE, sumE, nnz);
  k_weighted_accum<<<waveBlocks, 256, 0, stream>>>(X0, vertex, edges, logit, sumE,
                                                   Xe, nnz, 0);

  // 5-8) vertex-side attention on relu(Xe): softmax over vertex segments, out = segsum
  k_logits_max<<<waveBlocks, 256, 0, stream>>>(Xe, edges, vertex, ntype, vertex,
                                               att_v, logit, maxV, nnz, 1);
  k_expsum<<<nhBlocks, 256, 0, stream>>>(vertex, logit, maxV, sumV, nnz);
  k_weighted_accum<<<waveBlocks, 256, 0, stream>>>(Xe, edges, vertex, logit, sumV,
                                                   out, nnz, 1);
}
